// Seq2SeqGRU_42855183679989
// MI455X (gfx1250) — matrix operand order repaired and checked by reading, not compiled, not run on hardware
//
#include <hip/hip_runtime.h>

#define BATCH 256
#define TSEQ  512
#define EIN   64
#define HDIM  512
#define HZ    96

typedef __attribute__((ext_vector_type(16))) __bf16 v16bf;
typedef __attribute__((ext_vector_type(8)))  __bf16 v8bf;
typedef __attribute__((ext_vector_type(8)))  float  v8f;

__device__ __forceinline__ unsigned short f2bf_bits(float f) {
  union { float f; unsigned u; } v; v.f = f;
  unsigned r = v.u + 0x7FFFu + ((v.u >> 16) & 1u);   // round-to-nearest-even
  return (unsigned short)(r >> 16);
}

__global__ void cvt_bf16_kernel(const float* __restrict__ src,
                                unsigned short* __restrict__ dst, int n) {
  int i = blockIdx.x * blockDim.x + threadIdx.x;
  int stride = gridDim.x * blockDim.x;
  for (; i < n; i += stride) dst[i] = f2bf_bits(src[i]);
}

__device__ __forceinline__ v8f wmma_bf16(v16bf a, v16bf b, v8f c) {
  // (neg_a, A, neg_b, B, c_mod, C, reuse_a, reuse_b)
  return __builtin_amdgcn_wmma_f32_16x16x32_bf16(false, a, false, b, (short)0, c,
                                                 false, false);
}

// Launder a 32-bit element offset so LICM cannot hoist the weight loads out of
// the time loop (prevents the spill-to-scratch pathology) while keeping the
// base pointer's global-address-space provenance (emits global_load_b128, not
// flat_load_b128 which would charge DScnt too).
__device__ __forceinline__ unsigned opaque_off(unsigned v) {
  asm volatile("" : "+v"(v));
  return v;
}

// A-matrix 16x32 bf16 fragment: lane m=l16, half hf: K = hf*8..+7 and 16+hf*8..+7
__device__ __forceinline__ v16bf load_frag_a(const unsigned short* p) {
  v8bf lo = *(const v8bf*)p;
  v8bf hi = *(const v8bf*)(p + 16);
  return __builtin_shufflevector(lo, hi, 0,1,2,3,4,5,6,7,8,9,10,11,12,13,14,15);
}
// B-matrix 32x16 bf16 fragment: lane n=l16, half hf: K = hf*8..+7 and 16+hf*8..+7
__device__ __forceinline__ v16bf load_frag_b(const unsigned short* p) {
  v8bf lo = *(const v8bf*)p;
  v8bf hi = *(const v8bf*)(p + 16);
  return __builtin_shufflevector(lo, hi, 0,1,2,3,4,5,6,7,8,9,10,11,12,13,14,15);
}

__device__ __forceinline__ float sigmoidf_(float x) { return 1.0f / (1.0f + __expf(-x)); }

__global__ void __launch_bounds__(1024, 1)
gru_seq2seq_kernel(const float* __restrict__ X,            // (B,T,E) fp32 (for x0)
                   const unsigned short* __restrict__ Xbf, // (B,T,E) bf16
                   const unsigned short* __restrict__ WihE,// (3H,E)  bf16
                   const unsigned short* __restrict__ WhhE,// (3H,H)  bf16
                   const float* __restrict__ bihE, const float* __restrict__ bhhE,
                   const float* __restrict__ WihD,         // (3H,1)  fp32
                   const unsigned short* __restrict__ WhhD,// (3H,H)  bf16
                   const float* __restrict__ bihD, const float* __restrict__ bhhD,
                   const float* __restrict__ Wfc, const float* __restrict__ bfc,
                   float* __restrict__ out)
{
  __shared__ unsigned short h_bf[16][HDIM];      // bf16 hidden (WMMA A feed)
  __shared__ float          h_f32[16][HDIM + 1]; // exact hidden (padded)
  __shared__ unsigned short x_bf[16][EIN];       // bf16 x_t tile
  __shared__ float          x_s[16];             // decoder scalar input
  __shared__ float          red[16][17];         // W_fc dot reduction

  const int tid  = threadIdx.x;
  const int wave = tid >> 5;       // 32 waves: wave == owned column tile
  const int lane = tid & 31;
  const int hf   = lane >> 4;      // half-wave select
  const int l16  = lane & 15;
  const int b0   = blockIdx.x * 16;
  const int cb   = wave * 16;      // owned h-column base
  const int col  = cb + l16;

  for (int i = tid; i < 16 * HDIM; i += 1024) {
    int r = i >> 9, c = i & (HDIM - 1);
    h_f32[r][c] = 0.0f;
    h_bf[r][c]  = 0;
  }
  __syncthreads();

  // per-wave weight row element-offsets (K-contiguous rows)
  const unsigned offW = (unsigned)((cb + l16) * EIN  + hf * 8);
  const unsigned offU = (unsigned)((cb + l16) * HDIM + hf * 8);

  // encoder biases for this wave's columns (L2-resident scalars)
  const float ebr  = bihE[col]            + bhhE[col];
  const float ebz  = bihE[HDIM + col]     + bhhE[HDIM + col];
  const float ebgn = bihE[2 * HDIM + col];
  const float ebhn = bhhE[2 * HDIM + col];

  // ----------------------------- encoder -----------------------------
  for (int t = 0; t < TSEQ; ++t) {
    // stage x_t tile (1024 threads -> 1 element each)
    {
      int r = tid >> 6, c = tid & 63;
      x_bf[r][c] = Xbf[(size_t)(b0 + r) * (TSEQ * EIN) + t * EIN + c];
    }
    __syncthreads();

    // offsets re-laundered every iteration: loads stay in-loop, global
    const unsigned short* Wr = WihE + opaque_off(offW);
    const unsigned short* Wz = Wr + (size_t)HDIM * EIN;
    const unsigned short* Wn = Wz + (size_t)HDIM * EIN;
    const unsigned short* Ur = WhhE + opaque_off(offU);
    const unsigned short* Uz = Ur + (size_t)HDIM * HDIM;
    const unsigned short* Un = Uz + (size_t)HDIM * HDIM;

    v8f accr = {}, accz = {}, accgin = {}, accghn = {};

    // gi = x @ W_ih^T  (K = 64 -> 2 k-tiles)
#pragma unroll
    for (int k = 0; k < 2; ++k) {
      v16bf a = load_frag_a(&x_bf[l16][k * 32 + hf * 8]);
      accr   = wmma_bf16(a, load_frag_b(Wr + k * 32), accr);
      accz   = wmma_bf16(a, load_frag_b(Wz + k * 32), accz);
      accgin = wmma_bf16(a, load_frag_b(Wn + k * 32), accgin);
    }

    // gh = h @ W_hh^T (K = 512 -> 16 k-tiles), B pipelined 2 k-steps ahead
    {
      v16bf brb[2], bzb[2], bnb[2];
      brb[0] = load_frag_b(Ur);      bzb[0] = load_frag_b(Uz);      bnb[0] = load_frag_b(Un);
      brb[1] = load_frag_b(Ur + 32); bzb[1] = load_frag_b(Uz + 32); bnb[1] = load_frag_b(Un + 32);
#pragma unroll
      for (int k = 0; k < 16; ++k) {
        const int cur = k & 1;
        v16bf a  = load_frag_a(&h_bf[l16][k * 32 + hf * 8]);
        v16bf br = brb[cur], bz = bzb[cur], bn = bnb[cur];
        if (k < 14) {                       // issue k+2 loads before k's WMMAs
          brb[cur] = load_frag_b(Ur + (k + 2) * 32);
          bzb[cur] = load_frag_b(Uz + (k + 2) * 32);
          bnb[cur] = load_frag_b(Un + (k + 2) * 32);
        }
        accr   = wmma_bf16(a, br, accr);
        accz   = wmma_bf16(a, bz, accz);
        accghn = wmma_bf16(a, bn, accghn);
      }
    }

    float hnew[8];
#pragma unroll
    for (int i = 0; i < 8; ++i) {
      const int m = i + 8 * hf;             // C/D layout: VGPR i, halves -> M
      float r = sigmoidf_(accr[i] + ebr);
      float z = sigmoidf_(accz[i] + ebz);
      float n = tanhf(accgin[i] + ebgn + r * (accghn[i] + ebhn));
      hnew[i] = (1.0f - z) * n + z * h_f32[m][col];
    }
    __syncthreads();                        // all reads of old h complete
#pragma unroll
    for (int i = 0; i < 8; ++i) {
      const int m = i + 8 * hf;
      h_f32[m][col] = hnew[i];
      h_bf[m][col]  = f2bf_bits(hnew[i]);
    }
    __syncthreads();
  }

  // x0 = X[:, T-1, E-1]
  if (tid < 16)
    x_s[tid] = X[(size_t)(b0 + tid) * (TSEQ * EIN) + (TSEQ - 1) * EIN + (EIN - 1)];
  __syncthreads();

  // decoder biases / input weights for this wave's columns
  const float dwr  = WihD[col], dwz = WihD[HDIM + col], dwn = WihD[2 * HDIM + col];
  const float dbr  = bihD[col]        + bhhD[col];
  const float dbz  = bihD[HDIM + col] + bhhD[HDIM + col];
  const float dbgn = bihD[2 * HDIM + col];
  const float dbhn = bhhD[2 * HDIM + col];

  // ----------------------------- decoder -----------------------------
  for (int d = 0; d < HZ; ++d) {
    const unsigned short* Ur = WhhD + opaque_off(offU);
    const unsigned short* Uz = Ur + (size_t)HDIM * HDIM;
    const unsigned short* Un = Uz + (size_t)HDIM * HDIM;

    v8f accr = {}, accz = {}, accghn = {};
    {
      v16bf brb[2], bzb[2], bnb[2];
      brb[0] = load_frag_b(Ur);      bzb[0] = load_frag_b(Uz);      bnb[0] = load_frag_b(Un);
      brb[1] = load_frag_b(Ur + 32); bzb[1] = load_frag_b(Uz + 32); bnb[1] = load_frag_b(Un + 32);
#pragma unroll
      for (int k = 0; k < 16; ++k) {
        const int cur = k & 1;
        v16bf a  = load_frag_a(&h_bf[l16][k * 32 + hf * 8]);
        v16bf br = brb[cur], bz = bzb[cur], bn = bnb[cur];
        if (k < 14) {
          brb[cur] = load_frag_b(Ur + (k + 2) * 32);
          bzb[cur] = load_frag_b(Uz + (k + 2) * 32);
          bnb[cur] = load_frag_b(Un + (k + 2) * 32);
        }
        accr   = wmma_bf16(a, br, accr);
        accz   = wmma_bf16(a, bz, accz);
        accghn = wmma_bf16(a, bn, accghn);
      }
    }

    float hnew[8];
#pragma unroll
    for (int i = 0; i < 8; ++i) {
      const int m = i + 8 * hf;
      const float xm = x_s[m];
      float r = sigmoidf_(accr[i] + xm * dwr + dbr);
      float z = sigmoidf_(accz[i] + xm * dwz + dbz);
      float n = tanhf(xm * dwn + dbgn + r * (accghn[i] + dbhn));
      hnew[i] = (1.0f - z) * n + z * h_f32[m][col];
    }
    __syncthreads();
#pragma unroll
    for (int i = 0; i < 8; ++i) {
      const int m = i + 8 * hf;
      h_f32[m][col] = hnew[i];
      h_bf[m][col]  = f2bf_bits(hnew[i]);
    }
    __syncthreads();

    // pred = h_new @ W_fc^T + b_fc  (fp32 dot, 16 lanes per row)
    if (tid < 256) {
      const int row = tid >> 4, part = tid & 15;
      float s = 0.0f;
      for (int k = part; k < HDIM; k += 16) s += h_f32[row][k] * Wfc[k];
      red[row][part] = s;
    }
    __syncthreads();
    if (tid < 16) {
      float p = bfc[0];
#pragma unroll
      for (int j = 0; j < 16; ++j) p += red[tid][j];
      out[(size_t)(b0 + tid) * HZ + d] = p;
      x_s[tid] = p;                          // feed back as next decoder input
    }
    __syncthreads();
  }
}

extern "C" void kernel_launch(void* const* d_in, const int* in_sizes, int n_in,
                              void* d_out, int out_size, void* d_ws, size_t ws_size,
                              hipStream_t stream) {
  (void)in_sizes; (void)n_in; (void)out_size; (void)ws_size;
  const float* X      = (const float*)d_in[0];
  // d_in[1] = horizon (fixed 96, compile-time)
  const float* WihE_f = (const float*)d_in[2];
  const float* WhhE_f = (const float*)d_in[3];
  const float* bihE   = (const float*)d_in[4];
  const float* bhhE   = (const float*)d_in[5];
  const float* WihD   = (const float*)d_in[6];
  const float* WhhD_f = (const float*)d_in[7];
  const float* bihD   = (const float*)d_in[8];
  const float* bhhD   = (const float*)d_in[9];
  const float* Wfc    = (const float*)d_in[10];
  const float* bfc    = (const float*)d_in[11];
  float* out = (float*)d_out;

  char* ws = (char*)d_ws;
  unsigned short* Xbf   = (unsigned short*)(ws);                       // 16 MiB
  unsigned short* WihEb = (unsigned short*)(ws + 16777216);            // 192 KiB
  unsigned short* WhhEb = (unsigned short*)(ws + 16777216 + 262144);   // 1.5 MiB
  unsigned short* WhhDb = (unsigned short*)(ws + 16777216 + 262144 + 1572864);

  cvt_bf16_kernel<<<1024, 256, 0, stream>>>(X,      Xbf,   BATCH * TSEQ * EIN);
  cvt_bf16_kernel<<<128,  256, 0, stream>>>(WihE_f, WihEb, 3 * HDIM * EIN);
  cvt_bf16_kernel<<<512,  256, 0, stream>>>(WhhE_f, WhhEb, 3 * HDIM * HDIM);
  cvt_bf16_kernel<<<512,  256, 0, stream>>>(WhhD_f, WhhDb, 3 * HDIM * HDIM);

  gru_seq2seq_kernel<<<BATCH / 16, 1024, 0, stream>>>(
      X, Xbf, WihEb, WhhEb, bihE, bhhE, WihD, WhhDb, bihD, bhhD, Wfc, bfc, out);
}
